// ConstrainedKNNRetriever_80616536145988
// MI455X (gfx1250) — compile-verified, matrix operand order checked by reading, not run on hardware
//
#include <hip/hip_runtime.h>

// ---------------------------------------------------------------------------
// Constrained KNN retrieval for MI455X (gfx1250, wave32, WMMA).
// T=512 tokens, D=768, N=200000 candidates.
//   - features (614MB f32) streamed from HBM exactly once (26us floor @23.3TB/s)
//   - GEMM (157 GFLOP) via v_wmma_f32_16x16x32_bf16, f32 accumulate
//   - h-tile staged with GLOBAL_LOAD_ASYNC_TO_LDS_B128 (ASYNCcnt), double-
//     buffered single-barrier pipeline (issue slot after barrier)
//   - mask logic collapsed to token flags via tiny label histograms
//   - argmax via packed (orderedKey<<32 | ~idx) u64 atomic max (ties -> min idx)
// ---------------------------------------------------------------------------

#define T_TOK  512
#define DDIM   768
#define NCAND  200000
#define CB     64              // candidates per workgroup
#define NWG    (NCAND / CB)    // 3125
#define KSTEP  32
#define NSTEPS (DDIM / KSTEP)  // 24
#define NEGV   -1e30f

typedef __attribute__((ext_vector_type(16))) __bf16 v16bf;
typedef __attribute__((ext_vector_type(4)))  __bf16 v4bf;
typedef __attribute__((ext_vector_type(8)))  float  v8f;
typedef __attribute__((ext_vector_type(4)))  int    v4i;

union Frag { v16bf v; uint4 q[2]; };

#if defined(__AMDGCN__) && __has_builtin(__builtin_amdgcn_global_load_async_to_lds_b128)
#define USE_ASYNC 1
#else
#define USE_ASYNC 0
#endif

#if USE_ASYNC
typedef __attribute__((address_space(1))) v4i v4i_g;   // global int4
typedef __attribute__((address_space(3))) v4i v4i_l;   // LDS int4
#endif

__device__ __forceinline__ void cpAsync16(void* lds, const void* gptr) {
#if USE_ASYNC
  __builtin_amdgcn_global_load_async_to_lds_b128((v4i_g*)gptr, (v4i_l*)lds, 0, 0);
#else
  *(uint4*)lds = *(const uint4*)gptr;
#endif
}

__device__ __forceinline__ void waitAsync0() {
#if USE_ASYNC
#if __has_builtin(__builtin_amdgcn_s_wait_asynccnt)
  __builtin_amdgcn_s_wait_asynccnt(0);
#else
  asm volatile("s_wait_asynccnt 0" ::: "memory");
#endif
#endif
}

__device__ __forceinline__ unsigned orderKey(float f) {
  unsigned b = __float_as_uint(f);
  return (b & 0x80000000u) ? ~b : (b | 0x80000000u);
}

// --------------------------- kernel 1: zero ws ------------------------------
__global__ void k_zero(unsigned* __restrict__ hist, unsigned long long* __restrict__ best) {
  int i = blockIdx.x * 256 + threadIdx.x;
  if (i < 8384) hist[i] = 0u;          // 64 phone + 128 sym + 8192 pair
  if (i < T_TOK) best[i] = 0ULL;
}

// ----------------------- kernel 2: normalize h -> bf16 ----------------------
__global__ void k_hnorm(const float* __restrict__ h, __bf16* __restrict__ hN) {
  __shared__ float red[256];
  int t = blockIdx.x, tid = threadIdx.x;
  const float* row = h + (size_t)t * DDIM;
  float v0 = row[tid], v1 = row[tid + 256], v2 = row[tid + 512];
  red[tid] = v0 * v0 + v1 * v1 + v2 * v2;
  __syncthreads();
  for (int o = 128; o > 0; o >>= 1) {
    if (tid < o) red[tid] += red[tid + o];
    __syncthreads();
  }
  float inv = 1.0f / (sqrtf(red[0]) + 1e-8f);
  __bf16* orow = hN + (size_t)t * DDIM;
  orow[tid]       = (__bf16)(v0 * inv);
  orow[tid + 256] = (__bf16)(v1 * inv);
  orow[tid + 512] = (__bf16)(v2 * inv);
}

// ----------------------- kernel 3: label histograms -------------------------
__global__ void k_hist(const int* __restrict__ g, const int* __restrict__ pp,
                       const int* __restrict__ sp, const int* __restrict__ tg,
                       unsigned* __restrict__ phoneCnt, unsigned* __restrict__ symCnt,
                       unsigned* __restrict__ pairCnt) {
  int i = blockIdx.x * 256 + threadIdx.x;
  if (i >= NCAND) return;
  if (g[i] == tg[0]) {
    int p = pp[i], s = sp[i];
    atomicAdd(&phoneCnt[p], 1u);
    atomicAdd(&symCnt[s], 1u);
    atomicAdd(&pairCnt[p * 128 + s], 1u);
  }
}

// ----------------------- kernel 4: per-token flags --------------------------
__global__ void k_flags(const int* __restrict__ phones, const int* __restrict__ symbols,
                        const int* __restrict__ kPtr,
                        const unsigned* __restrict__ phoneCnt,
                        const unsigned* __restrict__ symCnt,
                        const unsigned* __restrict__ pairCnt,
                        unsigned* __restrict__ flags) {
  int t = threadIdx.x;  // 512 threads, 1 block
  int p = phones[t], s = symbols[t];
  unsigned kk = (unsigned)kPtr[0];
  unsigned usePhone = (phoneCnt[p] >= kk) ? 1u : 0u;
  unsigned c2 = usePhone ? pairCnt[p * 128 + s] : symCnt[s];
  unsigned useSym = (s >= 0 && c2 >= kk) ? 1u : 0u;
  flags[t] = usePhone | (useSym << 1);
}

// -------------------- kernel 5: WMMA GEMM + masked argmax -------------------
// 512 threads = 16 wave32; wave w: cand sub-block cb=w&3 (16 cands),
// token quarter tq=w>>2 (128 tokens = 8 WMMA tiles). WG covers 64 cands x 512 tok.
// Double-buffered LDS, one barrier + one asynccnt wait per K-step.
__global__ __launch_bounds__(512, 1)
void k_simknn(const float* __restrict__ feats, const __bf16* __restrict__ hN,
              const int* __restrict__ genders, const int* __restrict__ phones_pool,
              const int* __restrict__ symbols_pool, const unsigned* __restrict__ flags,
              const int* __restrict__ phones, const int* __restrict__ symbols,
              const int* __restrict__ tg, unsigned long long* __restrict__ best) {
  __shared__ __align__(16) unsigned char sH[2][T_TOK * 80]; // 512 x 32 bf16, 80B stride (pad)
  __shared__ __align__(16) unsigned char sF[2][CB * 80];    // 64 x 32 bf16, 80B stride
  __shared__ float sSum[CB];
  __shared__ unsigned sCand[CB];
  __shared__ unsigned sTok[T_TOK];
  __shared__ unsigned long long sBest[T_TOK];

  const int tid  = threadIdx.x;
  const int lane = tid & 31;
  const int w    = tid >> 5;
  const int cb   = w & 3;
  const int tq   = w >> 2;
  const int lm   = lane & 15;
  const int hi   = (lane & 16) ? 16 : 0;   // K-half select per A/B layout
  const int candBase = blockIdx.x * CB;

  // ---- prologue: init LDS state ----
  sBest[tid] = 0ULL;
  if (tid < CB) {
    int n = candBase + tid;
    unsigned info = (genders[n] == tg[0]) ? 1u : 0u;
    info |= ((unsigned)phones_pool[n] & 0x7fu) << 1;
    info |= ((unsigned)symbols_pool[n] & 0xffu) << 8;
    sCand[tid] = info;
    sSum[tid] = 0.0f;
  }
  sTok[tid] = ((unsigned)phones[tid] & 0xffu) |
              (((unsigned)symbols[tid] & 0xffu) << 8) | (flags[tid] << 16);

  v8f acc[8] = {};
  float ss = 0.0f;

  const int fc = tid >> 3;        // candidate 0..63 staged by this thread
  const int fk = (tid & 7) * 4;   // 4 f32 elems within the 32-wide K step
  const float* fsrc = feats + (size_t)(candBase + fc) * DDIM + fk;
  const __bf16* hRow = hN + (size_t)tid * DDIM;   // one h row per thread

  // ---- prologue staging: feature tile 0 + async h tile 0 ----
  {
    float4 x = *(const float4*)fsrc;
    ss += x.x * x.x + x.y * x.y + x.z * x.z + x.w * x.w;
    v4bf bv;
    bv.x = (__bf16)x.x; bv.y = (__bf16)x.y; bv.z = (__bf16)x.z; bv.w = (__bf16)x.w;
    *(v4bf*)(&sF[0][fc * 80 + fk * 2]) = bv;
  }
  {
    const uint4* g = (const uint4*)hRow;
    unsigned char* d = &sH[0][tid * 80];
    cpAsync16(d, g); cpAsync16(d + 16, g + 1);
    cpAsync16(d + 32, g + 2); cpAsync16(d + 48, g + 3);
  }
  float4 xa = *(const float4*)(fsrc + KSTEP);   // feature data for step 1

  for (int i = 0; i < NSTEPS; ++i) {
    const int b = i & 1;
    waitAsync0();      // h tile for step i has landed in LDS
    __syncthreads();   // everyone done computing step i-1; buffers b ready

    // ---- issue slot: stage step i+1 into buffers b^1 (overlaps WMMA below)
    if (i + 1 < NSTEPS) {
      const int k0n = (i + 1) * KSTEP;
      const uint4* g = (const uint4*)(hRow + k0n);
      unsigned char* d = &sH[b ^ 1][tid * 80];
      cpAsync16(d, g); cpAsync16(d + 16, g + 1);
      cpAsync16(d + 32, g + 2); cpAsync16(d + 48, g + 3);

      ss += xa.x * xa.x + xa.y * xa.y + xa.z * xa.z + xa.w * xa.w;
      v4bf bv;
      bv.x = (__bf16)xa.x; bv.y = (__bf16)xa.y; bv.z = (__bf16)xa.z; bv.w = (__bf16)xa.w;
      *(v4bf*)(&sF[b ^ 1][fc * 80 + fk * 2]) = bv;

      if (i + 2 < NSTEPS) xa = *(const float4*)(fsrc + (i + 2) * KSTEP);
    }

    // ---- compute: B fragment (K=32 x N=16) for this wave's cand sub-block;
    // K layout mirrors A: lanes 0-15 get K[0..7],[16..23]; lanes 16-31 K[8..15],[24..31]
    Frag bfr;
    {
      const unsigned char* p = &sF[b][(cb * 16 + lm) * 80 + hi];
      bfr.q[0] = *(const uint4*)p;
      bfr.q[1] = *(const uint4*)(p + 32);
    }
#pragma unroll
    for (int j = 0; j < 8; ++j) {
      Frag a;
      const unsigned char* p = &sH[b][(tq * 128 + j * 16 + lm) * 80 + hi];
      a.q[0] = *(const uint4*)p;
      a.q[1] = *(const uint4*)(p + 32);
      acc[j] = __builtin_amdgcn_wmma_f32_16x16x32_bf16(
          false, a.v, false, bfr.v, (short)0, acc[j], false, false);
    }
  }

  // ---- per-candidate inverse norm ----
  atomicAdd(&sSum[fc], ss);
  __syncthreads();

  const int myCand = cb * 16 + lm;
  const float invn = 1.0f / (sqrtf(sSum[myCand]) + 1e-8f);
  const unsigned ci = sCand[myCand];
  const int cBase = (int)(ci & 1u);
  const int cp = (int)((ci >> 1) & 0x7fu);
  const int cs = (int)((ci >> 8) & 0xffu);
  const unsigned gIdx = (unsigned)(candBase + myCand);

  // ---- mask + argmax over this WG's 64 candidates ----
#pragma unroll
  for (int j = 0; j < 8; ++j) {
    const int tokTile = tq * 128 + j * 16;
#pragma unroll
    for (int r = 0; r < 8; ++r) {
      const int tok = tokTile + ((lane & 16) ? (r + 8) : r);
      const unsigned ti = sTok[tok];
      const bool ok = cBase &&
                      (!(ti & 0x10000u) || cp == (int)(ti & 0xffu)) &&
                      (!(ti & 0x20000u) || cs == (int)((ti >> 8) & 0xffu));
      float v = ok ? acc[j][r] * invn : NEGV;
      unsigned vi = gIdx;
#pragma unroll
      for (int m = 1; m < 16; m <<= 1) {   // reduce within each 16-lane half
        float ov = __shfl_xor(v, m, 32);
        unsigned oi = (unsigned)__shfl_xor((int)vi, m, 32);
        if (ov > v || (ov == v && oi < vi)) { v = ov; vi = oi; }
      }
      if (lm == 0) {
        unsigned long long pk =
            ((unsigned long long)orderKey(v) << 32) | (unsigned long long)(~vi);
        atomicMax(&sBest[tok], pk);
      }
    }
  }
  __syncthreads();
  atomicMax(&best[tid], sBest[tid]);   // one global u64 atomic per token per WG
}

// ----------------------- kernel 6: gather winners ---------------------------
__global__ void k_gather(const float* __restrict__ feats,
                         const unsigned long long* __restrict__ best,
                         float* __restrict__ out) {
  int t = blockIdx.x, tid = threadIdx.x;
  unsigned long long b = best[t];
  unsigned idx = ~((unsigned)b);
  if (idx >= (unsigned)NCAND) idx = 0;
  const float* src = feats + (size_t)idx * DDIM;
  float* dst = out + (size_t)t * DDIM;
  dst[tid] = src[tid];
  dst[tid + 256] = src[tid + 256];
  dst[tid + 512] = src[tid + 512];
}

// ---------------------------------------------------------------------------
extern "C" void kernel_launch(void* const* d_in, const int* in_sizes, int n_in,
                              void* d_out, int out_size, void* d_ws, size_t ws_size,
                              hipStream_t stream) {
  const float* h            = (const float*)d_in[0];
  const float* feats        = (const float*)d_in[1];
  const int*   genders      = (const int*)d_in[2];
  const int*   phones_pool  = (const int*)d_in[3];
  const int*   symbols_pool = (const int*)d_in[4];
  const int*   phones       = (const int*)d_in[5];
  const int*   symbols      = (const int*)d_in[6];
  const int*   tg           = (const int*)d_in[7];
  const int*   kk           = (const int*)d_in[8];

  char* ws = (char*)d_ws;
  unsigned long long* best = (unsigned long long*)(ws + 0);      // 512 * 8B
  unsigned* flags = (unsigned*)(ws + 4096);                      // 512 * 4B
  unsigned* hist  = (unsigned*)(ws + 6144);                      // 8384 * 4B
  unsigned* phoneCnt = hist;
  unsigned* symCnt   = hist + 64;
  unsigned* pairCnt  = hist + 192;
  __bf16* hN = (__bf16*)(ws + 40960);                            // 512*768 bf16
  float* out = (float*)d_out;

  hipLaunchKernelGGL(k_zero,  dim3(33),           dim3(256), 0, stream, hist, best);
  hipLaunchKernelGGL(k_hnorm, dim3(T_TOK),        dim3(256), 0, stream, h, hN);
  hipLaunchKernelGGL(k_hist,  dim3((NCAND + 255) / 256), dim3(256), 0, stream,
                     genders, phones_pool, symbols_pool, tg, phoneCnt, symCnt, pairCnt);
  hipLaunchKernelGGL(k_flags, dim3(1),            dim3(512), 0, stream,
                     phones, symbols, kk, phoneCnt, symCnt, pairCnt, flags);
  hipLaunchKernelGGL(k_simknn, dim3(NWG),         dim3(512), 0, stream,
                     feats, hN, genders, phones_pool, symbols_pool, flags,
                     phones, symbols, tg, best);
  hipLaunchKernelGGL(k_gather, dim3(T_TOK),       dim3(256), 0, stream, feats, best, out);
}
